// BasicLayer_65953517797432
// MI455X (gfx1250) — matrix operand order repaired and checked by reading, not compiled, not run on hardware
//
#include <hip/hip_runtime.h>
#include <hip/hip_bf16.h>
#include <stdint.h>

// MI455X / gfx1250, wave32. bf16 WMMA 16x16x32, f32 accumulate.

typedef __attribute__((ext_vector_type(16))) __bf16 v16bf;
typedef __attribute__((ext_vector_type(8)))  float  v8f;

#define LOG2E 1.44269504088896340736f

__device__ __forceinline__ uint16_t f2bf(float f) {
  uint32_t u = __float_as_uint(f);
  uint32_t r = u + 0x7FFFu + ((u >> 16) & 1u);   // round-to-nearest-even
  return (uint16_t)(r >> 16);
}

// Pack two floats into a bf16x2 dword: rounding adds + one v_perm_b32.
__device__ __forceinline__ uint32_t pack2bf(float f0, float f1) {
  uint32_t a = __float_as_uint(f0) + 0x8000u;
  uint32_t b = __float_as_uint(f1) + 0x8000u;
  return __builtin_amdgcn_perm(b, a, 0x07060302u); // {b[31:16], a[31:16]}
}

union BFrag {              // 16 bf16 halves = 8 VGPRs (one WMMA A or B operand)
  v16bf    v;
  uint16_t u[16];
  uint32_t w[8];
  uint4    q[2];
};

__device__ __forceinline__ v8f wmma_bf16(const BFrag& a, const BFrag& b, v8f c) {
  return __builtin_amdgcn_wmma_f32_16x16x32_bf16(false, a.v, false, b.v,
                                                 (short)0, c, false, false);
}

// ---------------------------------------------------------------------------
// One-shot operand conversion kernels (memory-bound, tiny).
// ---------------------------------------------------------------------------
__global__ __launch_bounds__(256) void pack_bf16_kernel(
    const float* __restrict__ src, uint32_t* __restrict__ dst, int npairs) {
  int i = blockIdx.x * 256 + threadIdx.x;
  if (i < npairs) {
    uint32_t a = __float_as_uint(src[2 * i]);
    uint32_t b = __float_as_uint(src[2 * i + 1]);
    a += 0x7FFFu + ((a >> 16) & 1u);
    b += 0x7FFFu + ((b >> 16) & 1u);
    dst[i] = __builtin_amdgcn_perm(b, a, 0x07060302u);
  }
}

// x [C=256][N=4096] f32  ->  tok [N=4096][C=256] bf16 (coalesced bf16 writes)
__global__ __launch_bounds__(256) void transpose_x_kernel(
    const float* __restrict__ x, uint16_t* __restrict__ tok) {
  int id = blockIdx.x * 256 + threadIdx.x;       // 1,048,576 threads
  int c = id & 255, n = id >> 8;
  tok[(size_t)n * 256 + c] = f2bf(x[(size_t)c * 4096 + n]);
}

// ---------------------------------------------------------------------------
// Phase 1: QKV = tok @ qkv_w^T (all-bf16 operands, pre-converted).
// Per head h: Q[h][n][32] (scaled by log2(e)/sqrt(32)), K[h][n][32],
// Vt[h][32][n]. One wave = 16x16 tile; K = 256 = 8 WMMA.
// ---------------------------------------------------------------------------
__global__ __launch_bounds__(256) void qkv_kernel(
    const uint16_t* __restrict__ tok, const uint16_t* __restrict__ wB,
    uint16_t* __restrict__ Qb, uint16_t* __restrict__ Kb,
    uint16_t* __restrict__ Vt) {
  const int lane = threadIdx.x & 31;
  const int wave = threadIdx.x >> 5;
  const int task = blockIdx.x * 8 + wave;        // 256*48 = 12288 tiles
  const int m0 = (task & 255) << 4;
  const int o0 = (task >> 8) << 4;
  const int n  = lane & 15;
  const int hi = lane >> 4;

  v8f acc = {0.f,0.f,0.f,0.f,0.f,0.f,0.f,0.f};
  const uint16_t* ar = tok + (size_t)(m0 + n) * 256;
  const uint16_t* wr = wB + (size_t)(o0 + n) * 256 + hi * 16;

  for (int kc = 0; kc < 8; ++kc) {
    BFrag A;   // A(16x32): halves j -> K = hi*16 + j
    A.q[0] = *(const uint4*)(ar + kc * 32 + hi * 16);
    A.q[1] = *(const uint4*)(ar + kc * 32 + hi * 16 + 8);
    BFrag B;   // B(32x16): halves j -> K = hi*16 + j
    B.q[0] = *(const uint4*)(wr + kc * 32);
    B.q[1] = *(const uint4*)(wr + kc * 32 + 8);
    acc = wmma_bf16(A, B, acc);
  }

  const int s = o0 >> 8;                         // q/k/v select (tile-uniform)
  const int h = (o0 >> 5) & 7;
  const int d = (o0 & 31) + n;
  const float qscale = 0.17677669529663688f * LOG2E;
#pragma unroll
  for (int r = 0; r < 8; ++r) {
    const int token = m0 + r + 8 * hi;
    const float val = acc[r];
    if (s == 0)      Qb[((size_t)(h * 4096 + token)) * 32 + d] = f2bf(val * qscale);
    else if (s == 1) Kb[((size_t)(h * 4096 + token)) * 32 + d] = f2bf(val);
    else             Vt[((size_t)(h * 32 + d)) * 4096 + token] = f2bf(val);
  }
}

// ---------------------------------------------------------------------------
// Phase 2: fused masked flash attention (transposed formulation).
// S^T = K x Q^T (2 WMMA), integer-distance mask, streaming softmax in the
// log2 domain, O^T += V^T x P^T (2 WMMA). Wave-uniform block cull.
// ---------------------------------------------------------------------------
__global__ __launch_bounds__(256) void attn_kernel(
    const uint16_t* __restrict__ Qb, const uint16_t* __restrict__ Kb,
    const uint16_t* __restrict__ Vt, uint16_t* __restrict__ attn) {
  const int lane = threadIdx.x & 31;
  const int wave = threadIdx.x >> 5;
  const int task = blockIdx.x * 8 + wave;        // 8 heads * 256 tiles
  const int h  = task >> 8;
  const int q0 = (task & 255) << 4;
  const int n  = lane & 15;
  const int hi = lane >> 4;

  const int qz = q0 >> 8, qy = (q0 >> 4) & 15;   // tile-uniform; qx = n
  int dx2[8];
#pragma unroll
  for (int r = 0; r < 8; ++r) { int dx = n - (8 * hi + r); dx2[r] = dx * dx; }

  BFrag Qt;  // Q^T B-frag (loop invariant): halves 0..7 -> d = hi*8 + j ; 8..15 -> d = 16+hi*8+j
  {
    const uint4* qp = (const uint4*)(Qb + ((size_t)(h * 4096 + q0 + n)) * 32 + hi * 8);
    Qt.q[0] = qp[0]; Qt.q[1] = qp[2];
  }

  v8f o0v = {0.f,0.f,0.f,0.f,0.f,0.f,0.f,0.f};   // O^T rows d = r+8*hi
  v8f o1v = {0.f,0.f,0.f,0.f,0.f,0.f,0.f,0.f};   // O^T rows d = 16+r+8*hi
  float m_run = -1e30f, l_run = 0.f;

  for (int k0 = 0; k0 < 4096; k0 += 32) {
    const int dz  = qz - (k0 >> 8);
    const int kyb = (k0 >> 4) & 15;
    const int dy0 = qy - kyb, dy1 = qy - (kyb + 1);
    const int c0 = dz * dz + dy0 * dy0;
    const int c1 = dz * dz + dy1 * dy1;
    if (c0 >= 100 && c1 >= 100) continue;        // wave-uniform cull

    BFrag Ka, Kc;  // K A-frags: row m = key, K-dim = hd
    {
      const uint16_t* kr0 = Kb + ((size_t)(h * 4096 + k0 + n)) * 32;
      const uint16_t* kr1 = kr0 + 16 * 32;
      Ka.q[0] = *(const uint4*)(kr0 + hi * 8);
      Ka.q[1] = *(const uint4*)(kr0 + 16 + hi * 8);
      Kc.q[0] = *(const uint4*)(kr1 + hi * 8);
      Kc.q[1] = *(const uint4*)(kr1 + 16 + hi * 8);
    }
    const v8f z = {0.f,0.f,0.f,0.f,0.f,0.f,0.f,0.f};
    v8f s0 = wmma_bf16(Ka, Qt, z);               // rows: keys k0 + r+8*hi
    v8f s1 = wmma_bf16(Kc, Qt, z);               // rows: keys k0+16 + r+8*hi

    float p0[8], p1[8];
    float bm = -1e30f;
#pragma unroll
    for (int r = 0; r < 8; ++r) {
      float a = (c0 + dx2[r] < 100) ? s0[r] : -1e30f;
      float b = (c1 + dx2[r] < 100) ? s1[r] : -1e30f;
      p0[r] = a; p1[r] = b;
      bm = fmaxf(bm, fmaxf(a, b));
    }
    bm = fmaxf(bm, __shfl_xor(bm, 16, 32));
    const float m_new = fmaxf(m_run, bm);
    const float alpha = __builtin_amdgcn_exp2f(m_run - m_new);
    float lsum = 0.f;
#pragma unroll
    for (int r = 0; r < 8; ++r) {                // scores already * log2(e)
      float e0 = __builtin_amdgcn_exp2f(p0[r] - m_new);
      float e1 = __builtin_amdgcn_exp2f(p1[r] - m_new);
      p0[r] = e0; p1[r] = e1;
      lsum += e0 + e1;
    }
    lsum += __shfl_xor(lsum, 16, 32);
    l_run = l_run * alpha + lsum;
    m_run = m_new;
#pragma unroll
    for (int r = 0; r < 8; ++r) { o0v[r] *= alpha; o1v[r] *= alpha; }

    // Assemble P^T B-frag (K=32 keys x N=16 queries) from each lane's own packed scores.
    uint32_t pk0[4], pk1[4], x0[4], x1[4];
#pragma unroll
    for (int i = 0; i < 4; ++i) {
      pk0[i] = pack2bf(p0[2 * i], p0[2 * i + 1]);
      pk1[i] = pack2bf(p1[2 * i], p1[2 * i + 1]);
      x0[i] = __shfl_xor((int)pk0[i], 16, 32);
      x1[i] = __shfl_xor((int)pk1[i], 16, 32);
    }
    BFrag P;
#pragma unroll
    for (int i = 0; i < 4; ++i) {
      P.w[i]     = pk0[i];                       // keys hi*8 + 2i, 2i+1
      P.w[4 + i] = pk1[i];                       // keys 16 + hi*8 + 2i, ...
    }

    BFrag Va, Vb;  // V^T A-frags: row m = d, K-dim = keys (contiguous rows)
    {
      const uint16_t* vr0 = Vt + ((size_t)(h * 32 + n)) * 4096 + k0;
      const uint16_t* vr1 = Vt + ((size_t)(h * 32 + 16 + n)) * 4096 + k0;
      Va.q[0] = *(const uint4*)(vr0 + hi * 8);
      Va.q[1] = *(const uint4*)(vr0 + 16 + hi * 8);
      Vb.q[0] = *(const uint4*)(vr1 + hi * 8);
      Vb.q[1] = *(const uint4*)(vr1 + 16 + hi * 8);
    }
    o0v = wmma_bf16(Va, P, o0v);
    o1v = wmma_bf16(Vb, P, o1v);
  }

  const float inv = 1.0f / l_run;                // self-key always valid
  uint16_t* arow = attn + (size_t)(q0 + n) * 256 + h * 32;
#pragma unroll
  for (int r = 0; r < 8; ++r) {
    arow[r + 8 * hi]      = f2bf(o0v[r] * inv);
    arow[16 + r + 8 * hi] = f2bf(o1v[r] * inv);
  }
}

// ---------------------------------------------------------------------------
// Phase 3: out = attn @ proj_w^T + b, written as [C][N] (b c d h w).
// ---------------------------------------------------------------------------
__global__ __launch_bounds__(256) void proj_kernel(
    const uint16_t* __restrict__ attn, const uint16_t* __restrict__ wB,
    const float* __restrict__ bias, float* __restrict__ out) {
  const int lane = threadIdx.x & 31;
  const int wave = threadIdx.x >> 5;
  const int task = blockIdx.x * 8 + wave;        // 256*16 = 4096 tiles
  const int m0 = (task & 255) << 4;
  const int o0 = (task >> 8) << 4;
  const int n  = lane & 15;
  const int hi = lane >> 4;

  v8f acc = {0.f,0.f,0.f,0.f,0.f,0.f,0.f,0.f};
  const uint16_t* ar = attn + (size_t)(m0 + n) * 256;
  const uint16_t* wr = wB + (size_t)(o0 + n) * 256 + hi * 16;

  for (int kc = 0; kc < 8; ++kc) {
    BFrag A;
    A.q[0] = *(const uint4*)(ar + kc * 32 + hi * 16);
    A.q[1] = *(const uint4*)(ar + kc * 32 + hi * 16 + 8);
    BFrag B;
    B.q[0] = *(const uint4*)(wr + kc * 32);
    B.q[1] = *(const uint4*)(wr + kc * 32 + 8);
    acc = wmma_bf16(A, B, acc);
  }
  const float bv = bias[o0 + n];
#pragma unroll
  for (int r = 0; r < 8; ++r)
    out[(size_t)(o0 + n) * 4096 + m0 + r + 8 * hi] = acc[r] + bv;
}

// ---------------------------------------------------------------------------
extern "C" void kernel_launch(void* const* d_in, const int* in_sizes, int n_in,
                              void* d_out, int out_size, void* d_ws, size_t ws_size,
                              hipStream_t stream) {
  const float* x      = (const float*)d_in[0];   // [1,256,16,16,16]
  const float* qkv_w  = (const float*)d_in[1];   // [768,256]
  const float* proj_w = (const float*)d_in[2];   // [256,256]
  const float* proj_b = (const float*)d_in[3];   // [256]
  float* out = (float*)d_out;                    // [1,256,16,16,16]

  uint8_t* ws = (uint8_t*)d_ws;
  uint16_t* Qb    = (uint16_t*)(ws);                               // 2 MB
  uint16_t* Kb    = (uint16_t*)(ws + ((size_t)2 << 20));           // 2 MB
  uint16_t* Vt    = (uint16_t*)(ws + ((size_t)4 << 20));           // 2 MB
  uint16_t* tokB  = (uint16_t*)(ws + ((size_t)6 << 20));           // 2 MB
  uint16_t* attnB = tokB;                        // tok dead after phase 1
  uint16_t* qkvWB = (uint16_t*)(ws + ((size_t)8 << 20));           // 384 KB
  uint16_t* prjWB = (uint16_t*)(ws + ((size_t)8 << 20) + (512u << 10)); // 128 KB

  pack_bf16_kernel<<<384, 256, 0, stream>>>(qkv_w, (uint32_t*)qkvWB, 98304);
  pack_bf16_kernel<<<128, 256, 0, stream>>>(proj_w, (uint32_t*)prjWB, 32768);
  transpose_x_kernel<<<4096, 256, 0, stream>>>(x, tokB);

  qkv_kernel<<<1536, 256, 0, stream>>>(tokB, qkvWB, Qb, Kb, Vt);
  attn_kernel<<<256, 256, 0, stream>>>(Qb, Kb, Vt, attnB);
  proj_kernel<<<512, 256, 0, stream>>>(attnB, prjWB, proj_b, out);
}
